// INT8QuantizedLinear_35485019799671
// MI455X (gfx1250) — matrix operand order repaired and checked by reading, not compiled, not run on hardware
//
#include <hip/hip_runtime.h>
#include <stdint.h>

typedef __attribute__((ext_vector_type(16))) _Float16 v16h;
typedef __attribute__((ext_vector_type(8)))  _Float16 v8h;
typedef __attribute__((ext_vector_type(8)))  float    v8f;
typedef __attribute__((ext_vector_type(4)))  float    v4f;
typedef __attribute__((ext_vector_type(16))) int8_t   v16c;

#define IN_F  4096
#define OUT_F 11008
#define MTOT  8192           // 4 * 2048
#define NGRP  32             // IN_F / 128
#define BM    128
#define BN    128
#define BK    64
#define NKT   (IN_F / BK)    // 64 K-tiles
#define LDA   72             // padded LDS row stride in halves (144B: 16B-aligned chunks, bank stagger)

__global__ __launch_bounds__(256)
void int8_linear_wmma_f16(const float* __restrict__ X,
                          const int8_t* __restrict__ Wq,
                          const float* __restrict__ S,
                          const float* __restrict__ Bias,
                          float* __restrict__ Y)
{
    __shared__ alignas(16) _Float16 As[2][BM][LDA];   // x tile, f16
    __shared__ alignas(16) _Float16 Bs[2][BN][LDA];   // dequantized W tile, f16

    const int tid  = threadIdx.x;
    const int lane = tid & 31;
    const int wave = tid >> 5;
    const int wm   = wave & 1;     // 2 wave-rows of 64
    const int wn   = wave >> 1;    // 4 wave-cols of 32

    const int bm = blockIdx.y * BM;
    const int bn = blockIdx.x * BN;

    // staging decomposition: X tile 128x64 f32 (8 passes of float4), W tile 128x64 i8 (2 passes of 16B)
    const int xr = tid >> 4;            // 0..15
    const int xc = (tid & 15) << 2;     // 0..60, float4 chunk
    const int wr = tid >> 2;            // 0..63
    const int wc = (tid & 3) << 4;      // 0,16,32,48 byte chunk

    v4f   xv[8];
    v16c  wv[2];
    float sv[2];

    auto stage_load = [&](int kt) {     // global -> registers (no LDS dependence)
        const int k0 = kt * BK;
        #pragma unroll
        for (int p = 0; p < 8; ++p)
            xv[p] = *(const v4f*)(X + (size_t)(bm + p * 16 + xr) * IN_F + k0 + xc);
        #pragma unroll
        for (int p = 0; p < 2; ++p) {
            wv[p] = *(const v16c*)(Wq + (size_t)(bn + p * 64 + wr) * IN_F + k0 + wc);
            sv[p] = S[(size_t)(bn + p * 64 + wr) * NGRP + (kt >> 1)];  // group = (kt*64)/128
        }
        if (kt + 1 < NKT) {             // warm L2/WGP$ for the tile after next
            __builtin_prefetch(X  + (size_t)(bm + xr) * IN_F + k0 + BK + xc, 0, 1);
            __builtin_prefetch(Wq + (size_t)(bn + wr) * IN_F + k0 + BK + wc, 0, 1);
        }
    };

    auto stage_store = [&](int buf) {   // registers -> LDS with fused convert/dequant
        #pragma unroll
        for (int p = 0; p < 8; ++p) {
            _Float16* d = &As[buf][p * 16 + xr][xc];
            #pragma unroll
            for (int i = 0; i < 4; ++i) d[i] = (_Float16)xv[p][i];
        }
        #pragma unroll
        for (int p = 0; p < 2; ++p) {
            _Float16* d = &Bs[buf][p * 64 + wr][wc];
            #pragma unroll
            for (int i = 0; i < 16; ++i) d[i] = (_Float16)((float)wv[p][i] * sv[p]);
        }
    };

    v8f acc[4][2];
    #pragma unroll
    for (int mf = 0; mf < 4; ++mf)
        #pragma unroll
        for (int nf = 0; nf < 2; ++nf)
            acc[mf][nf] = (v8f){0.f, 0.f, 0.f, 0.f, 0.f, 0.f, 0.f, 0.f};

    auto compute = [&](int buf) {
        #pragma unroll
        for (int ks = 0; ks < BK; ks += 32) {
            // B frag: lane holds column n = lane%16; chunks K = kq..kq+7 and kq+16..kq+23, kq = ks + (lane<16?0:8)
            v16h bf[2];
            const int kqb = ks + ((lane & 16) ? 8 : 0);
            #pragma unroll
            for (int nf = 0; nf < 2; ++nf) {
                const _Float16* bp = &Bs[buf][wn * 32 + nf * 16 + (lane & 15)][kqb];
                v8h blo = *(const v8h*)bp;
                v8h bhi = *(const v8h*)(bp + 16);
                #pragma unroll
                for (int i = 0; i < 8; ++i) { bf[nf][i] = blo[i]; bf[nf][i + 8] = bhi[i]; }
            }
            // A frag: lane holds row m = lane%16; chunks K = kq..kq+7 and kq+16..kq+23, kq = ks + (lane<16?0:8)
            const int kqa = ks + ((lane & 16) ? 8 : 0);
            #pragma unroll
            for (int mf = 0; mf < 4; ++mf) {
                const _Float16* ap = &As[buf][wm * 64 + mf * 16 + (lane & 15)][kqa];
                v8h lo = *(const v8h*)ap;
                v8h hi = *(const v8h*)(ap + 16);
                v16h af;
                #pragma unroll
                for (int i = 0; i < 8; ++i) { af[i] = lo[i]; af[i + 8] = hi[i]; }
                #pragma unroll
                for (int nf = 0; nf < 2; ++nf)
                    acc[mf][nf] = __builtin_amdgcn_wmma_f32_16x16x32_f16(
                        false, af, false, bf[nf], (short)0, acc[mf][nf], false, false);
            }
        }
    };

    // software pipeline: load(k+1) || compute(k), then convert+store(k+1), barrier
    stage_load(0);
    stage_store(0);
    __syncthreads();

    for (int kt = 0; kt < NKT; ++kt) {
        const int  cur  = kt & 1;
        const bool more = (kt + 1) < NKT;
        if (more) stage_load(kt + 1);
        compute(cur);
        if (more) stage_store(cur ^ 1);
        __syncthreads();
    }

    // epilogue: match reference numerics: f16(acc) + f16(bias) -> f32
    const int nc = lane & 15;
    const int mo = (lane & 16) ? 8 : 0;
    #pragma unroll
    for (int mf = 0; mf < 4; ++mf) {
        const size_t mrow = (size_t)(bm + wm * 64 + mf * 16 + mo);
        #pragma unroll
        for (int nf = 0; nf < 2; ++nf) {
            const int ng = bn + wn * 32 + nf * 16 + nc;
            const _Float16 hb = (_Float16)Bias[ng];
            #pragma unroll
            for (int j = 0; j < 8; ++j) {
                _Float16 h = (_Float16)acc[mf][nf][j] + hb;
                Y[(mrow + (size_t)j) * (size_t)OUT_F + ng] = (float)h;
            }
        }
    }
}

extern "C" void kernel_launch(void* const* d_in, const int* in_sizes, int n_in,
                              void* d_out, int out_size, void* d_ws, size_t ws_size,
                              hipStream_t stream) {
    const float*  X  = (const float*)d_in[0];    // (4,2048,4096) f32
    const int8_t* Wq = (const int8_t*)d_in[1];   // (11008,4096) i8
    const float*  S  = (const float*)d_in[2];    // (11008,32) f32
    const float*  B  = (const float*)d_in[3];    // (11008,) f32
    float* Y = (float*)d_out;                    // (4,2048,11008) f32
    (void)in_sizes; (void)n_in; (void)out_size; (void)d_ws; (void)ws_size;

    dim3 grid(OUT_F / BN, MTOT / BM);            // 86 x 64 workgroups
    dim3 block(256);                             // 8 waves (wave32)
    int8_linear_wmma_f16<<<grid, block, 0, stream>>>(X, Wq, S, B, Y);
}
